// GroupConv_40381282517217
// MI455X (gfx1250) — matrix operand order repaired and checked by reading, not compiled, not run on hardware
//
#include <hip/hip_runtime.h>
#include <hip/hip_bf16.h>
#include <math.h>

// ---------------------------------------------------------------------------
// Problem constants
//   x:   [B=16, Cin=32, G=8, H=64, W=64] f32
//   out: [B=16, Cout=32, Gout=8, H=64, W=64] f32
//   Implicit GEMM: M = 256, N = 65536, Kpad = (Cin*Gin)*64 = 16384
// ---------------------------------------------------------------------------
#define NPTS   3136
#define MDIM   256
#define KPAD   16384
#define SAK    40        // LDS row stride (shorts): 80B, keeps b128 frags 16B-aligned
#define SBK    40

typedef __bf16 v16bf __attribute__((ext_vector_type(16)));
typedef float  v8f   __attribute__((ext_vector_type(8)));
typedef int    i32x4 __attribute__((vector_size(16)));   // gcc-style, matches builtin sig

#define AS1 __attribute__((address_space(1)))
#define AS3 __attribute__((address_space(3)))

#if defined(__has_builtin)
#  if __has_builtin(__builtin_amdgcn_global_load_async_to_lds_b128)
#    define HAVE_ASYNC_LDS 1
#  endif
#endif
#ifndef HAVE_ASYNC_LDS
#  define HAVE_ASYNC_LDS 0
#endif

#if HAVE_ASYNC_LDS
#  if __has_builtin(__builtin_amdgcn_s_wait_asynccnt)
#    define WAIT_ASYNC() __builtin_amdgcn_s_wait_asynccnt(0)
#  else
#    define WAIT_ASYNC() asm volatile("s_wait_asynccnt 0x0" ::: "memory")
#  endif
#else
#  define WAIT_ASYNC() ((void)0)
#endif

__device__ __forceinline__ unsigned short f2bf(float f) {
    union { float f; unsigned u; } c; c.f = f;
    unsigned u = c.u;
    unsigned r = u + 0x7FFFu + ((u >> 16) & 1u);   // round-to-nearest-even
    return (unsigned short)(r >> 16);
}

// A fragment (16x32 bf16): lane<16 -> row m, K 0..15 contiguous ;
// lane>=16 -> K 16..31.
__device__ __forceinline__ v16bf ldfragA(const unsigned short* p) {
    union { v16bf v; uint4 q[2]; } u;
    u.q[0] = *(const uint4*)(p);
    u.q[1] = *(const uint4*)(p + 8);
    return u.v;
}
// B fragment (32x16 bf16): lane<16 -> col n, K 0..15 contiguous ; lane>=16 -> K 16..31.
__device__ __forceinline__ v16bf ldfragB(const unsigned short* p) {
    union { v16bf v; uint4 q[2]; } u;
    u.q[0] = *(const uint4*)(p);
    u.q[1] = *(const uint4*)(p + 8);
    return u.v;
}

// ---------------------------------------------------------------------------
// Kernel 0a: zero packed weights (covers the K-padding slots)
// ---------------------------------------------------------------------------
__global__ void zero_fill_u4(uint4* p, int n) {
    int i = blockIdx.x * blockDim.x + threadIdx.x;
    if (i < n) p[i] = make_uint4(0u, 0u, 0u, 0u);
}

// ---------------------------------------------------------------------------
// Kernel 0b: f32 -> bf16 convert of x (one pass; removes per-stage cvt VALU
// work from the conv inner loop and halves L2 restage traffic)
// ---------------------------------------------------------------------------
__global__ void cvt_bf16(const float* __restrict__ x, unsigned short* __restrict__ xbf, int n4) {
    int i = blockIdx.x * blockDim.x + threadIdx.x;
    if (i < n4) {
        const float4 f = ((const float4*)x)[i];
        ushort4 o;
        o.x = f2bf(f.x); o.y = f2bf(f.y); o.z = f2bf(f.z); o.w = f2bf(f.w);
        ((ushort4*)xbf)[i] = o;
    }
}

// ---------------------------------------------------------------------------
// Kernel 1: SIREN hidden activations h2[p][32], one wave per point
// ---------------------------------------------------------------------------
__global__ void siren_hidden(const float* __restrict__ thetas,
                             const float* __restrict__ W1, const float* __restrict__ b1,
                             const float* __restrict__ W2, const float* __restrict__ b2,
                             float* __restrict__ h2out) {
    __shared__ float h1s[8][32];
    const int tid  = threadIdx.x;
    const int slot = tid >> 5;
    const int j    = tid & 31;
    const int p    = blockIdx.x * 8 + slot;       // 392 * 8 = 3136 exactly

    const int go  = p / 392;
    const int rem = p - go * 392;
    const int gi  = rem / 49;
    const int s   = rem - gi * 49;
    const int ky  = s / 7;
    const int kx  = s - ky * 7;

    const float th = thetas[go];
    const float ct = cosf(th), st = sinf(th);
    const float yy = (float)ky * (1.0f / 3.0f) - 1.0f;
    const float xx = (float)kx * (1.0f / 3.0f) - 1.0f;
    const float a0 =  ct * yy + st * xx;          // R(-theta) @ rel_pos
    const float a1 = -st * yy + ct * xx;
    float d = thetas[gi] - th;
    const float TWO_PI = 6.28318530717958647692f;
    d = d - floorf(d / TWO_PI) * TWO_PI;
    const float ag = d * (1.0f / 3.14159265358979323846f) - 1.0f;

    float z = a0 * W1[j] + a1 * W1[32 + j] + ag * W1[64 + j] + b1[j];
    h1s[slot][j] = sinf(10.0f * z);
    __syncthreads();

    float acc = b2[j];
    #pragma unroll
    for (int k = 0; k < 32; ++k) acc += h1s[slot][k] * W2[k * 32 + j];
    h2out[p * 32 + j] = sinf(10.0f * acc);
}

// ---------------------------------------------------------------------------
// Kernel 2: final SIREN layer + disk mask; pack bf16 weights as Wpk[M][K]
//   K = (ci*8 + gi)*64 + (ky*7 + kx)   (matches x channel order ci*8+gi)
//   M = go*32 + co                     (K contiguous -> async b128 staging)
// ---------------------------------------------------------------------------
__global__ void siren_pack(const float* __restrict__ thetas,
                           const float* __restrict__ W3, const float* __restrict__ b3,
                           const float* __restrict__ h2in,
                           unsigned short* __restrict__ Wpk) {
    __shared__ float h2s[32];
    const int tid = threadIdx.x;
    const int p   = blockIdx.x;
    if (tid < 32) h2s[tid] = h2in[p * 32 + tid];
    __syncthreads();

    const int go  = p / 392;
    const int rem = p - go * 392;
    const int gi  = rem / 49;
    const int s   = rem - gi * 49;
    const int ky  = s / 7;
    const int kx  = s - ky * 7;

    const float th = thetas[go];
    const float ct = cosf(th), st = sinf(th);
    const float yy = (float)ky * (1.0f / 3.0f) - 1.0f;
    const float xx = (float)kx * (1.0f / 3.0f) - 1.0f;
    const float a0 =  ct * yy + st * xx;
    const float a1 = -st * yy + ct * xx;
    const float maskf = (sqrtf(a0 * a0 + a1 * a1) <= 1.0f) ? 1.0f : 0.0f;

    #pragma unroll
    for (int jj = 0; jj < 4; ++jj) {
        const int m1024 = tid + (jj << 8);        // coalesced W3 reads
        float acc = b3[m1024];
        #pragma unroll
        for (int k = 0; k < 32; ++k) acc += h2s[k] * W3[k * 1024 + m1024];
        acc *= maskf;
        const int co = m1024 >> 5;
        const int ci = m1024 & 31;
        const int kk = (ci * 8 + gi) * 64 + s;
        Wpk[(size_t)(go * 32 + co) * KPAD + kk] = f2bf(acc);
    }
}

// ---------------------------------------------------------------------------
// Kernel 3: implicit-GEMM conv via v_wmma_f32_16x16x32_bf16
//   block tile 128(M) x 128(N), K-step 32; 8 waves, each a 16x128 strip
//   (8 WMMA tiles/wave/K-step); double-buffered LDS; A tile staged with
//   GLOBAL_LOAD_ASYNC_TO_LDS_B128 (ASYNCcnt), B tile via VGPRs (halo mask).
// ---------------------------------------------------------------------------
__device__ __forceinline__ void stage_tiles(unsigned short* __restrict__ sAbuf,
                                            unsigned short* __restrict__ sBbuf,
                                            int kb, int tid,
                                            const unsigned short* __restrict__ xbf,
                                            const unsigned short* __restrict__ Wpk,
                                            int mb, int nb) {
    // ---- A tile: 128 m x 32 k = 8KB, pure copy, 2 x async b128 per thread
    #pragma unroll
    for (int i = 0; i < 2; ++i) {
        const int lin = tid + (i << 8);          // 0..511
        const int m_l = lin >> 2;                // 0..127
        const int c   = lin & 3;                 // 8-k chunk
        const unsigned short* gp = Wpk + (size_t)(mb + m_l) * KPAD + kb + (c << 3);
        unsigned short*       lp = sAbuf + m_l * SAK + (c << 3);
#if HAVE_ASYNC_LDS
        __builtin_amdgcn_global_load_async_to_lds_b128((AS1 i32x4*)gp, (AS3 i32x4*)lp, 0, 0);
#else
        *(uint4*)lp = *(const uint4*)gp;
#endif
    }
    // ---- B tile: 128 n x 32 k, u16 gathers with halo masking
    const int n_l = tid & 127;
    const int k0  = tid >> 7;
    const int n_g = nb + n_l;
    const int bI  = n_g >> 12;
    const int hw  = n_g & 4095;
    const int h   = hw >> 6;
    const int w   = hw & 63;
    const unsigned short* xb = xbf + (size_t)bI * (256u * 4096u);
    #pragma unroll
    for (int i = 0; i < 16; ++i) {
        const int k_l  = k0 + (i << 1);
        const int k_g  = kb + k_l;
        const int cidx = k_g >> 6;               // ci*8+gi, constant within a K-step
        const int s    = k_g & 63;
        unsigned short v = 0;
        if (s < 49) {
            const int ky = s / 7;
            const int kx = s - ky * 7;
            const int ih = h + ky - 3;
            const int iw = w + kx - 3;
            if ((unsigned)ih < 64u && (unsigned)iw < 64u)
                v = xb[cidx * 4096 + ih * 64 + iw];
        }
        sBbuf[n_l * SBK + k_l] = v;
    }
}

__global__ void __launch_bounds__(256, 1)
conv_wmma(const unsigned short* __restrict__ xbf, const unsigned short* __restrict__ Wpk,
          const float* __restrict__ bias, float* __restrict__ out) {
    __shared__ __attribute__((aligned(16))) unsigned short sA[2][128 * SAK];
    __shared__ __attribute__((aligned(16))) unsigned short sB[2][128 * SBK];

    const int tid  = threadIdx.x;
    const int lane = tid & 31;
    const int wid  = tid >> 5;          // 8 M-waves, each 16 rows x 128 cols
    const int mb   = blockIdx.y << 7;   // 2 M-blocks
    const int nb   = blockIdx.x << 7;   // 512 N-blocks

    v8f acc[8] = {};

    stage_tiles(sA[0], sB[0], 0, tid, xbf, Wpk, mb, nb);
    WAIT_ASYNC();
    __syncthreads();

    const int arow = ((lane & 15) + (wid << 4)) * SAK + ((lane < 16) ? 0 : 16);
    const int kbB  = (lane < 16) ? 0 : 16;
    const int bcol = lane & 15;

    for (int it = 0; it < KPAD / 32; ++it) {
        const int cur = it & 1;
        if (it != KPAD / 32 - 1)
            stage_tiles(sA[cur ^ 1], sB[cur ^ 1], (it + 1) << 5, tid, xbf, Wpk, mb, nb);

        const v16bf a = ldfragA(&sA[cur][arow]);
        #pragma unroll
        for (int t = 0; t < 8; ++t) {
            const v16bf b = ldfragB(&sB[cur][((t << 4) + bcol) * SBK + kbB]);
            acc[t] = __builtin_amdgcn_wmma_f32_16x16x32_bf16(
                false, a, false, b, (short)0, acc[t], false, false);
        }
        WAIT_ASYNC();
        __syncthreads();
    }

    // Epilogue: C/D layout -> out[B][Cout][Gout][H][W], add bias[co]
    #pragma unroll
    for (int t = 0; t < 8; ++t) {
        const int n_g = nb + (t << 4) + (lane & 15);
        const int bI  = n_g >> 12;
        const int hw  = n_g & 4095;
        #pragma unroll
        for (int r = 0; r < 8; ++r) {
            const int m_g = mb + (wid << 4) + r + ((lane < 16) ? 0 : 8);
            const int go  = m_g >> 5;
            const int co  = m_g & 31;
            out[(size_t)(((bI << 5) + co) * 8 + go) * 4096u + hw] = acc[t][r] + bias[co];
        }
    }
}

// ---------------------------------------------------------------------------
// Host launcher
// ---------------------------------------------------------------------------
extern "C" void kernel_launch(void* const* d_in, const int* in_sizes, int n_in,
                              void* d_out, int out_size, void* d_ws, size_t ws_size,
                              hipStream_t stream) {
    const float* x    = (const float*)d_in[0];
    const float* gel  = (const float*)d_in[1];   // thetas, 8 floats
    const float* W1   = (const float*)d_in[2];
    const float* b1   = (const float*)d_in[3];
    const float* W2   = (const float*)d_in[4];
    const float* b2   = (const float*)d_in[5];
    const float* W3   = (const float*)d_in[6];
    const float* b3   = (const float*)d_in[7];
    const float* bias = (const float*)d_in[8];
    float* out = (float*)d_out;

    // workspace layout
    float*          h2  = (float*)d_ws;                                      // 3136*32 f32
    unsigned short* Wpk = (unsigned short*)((char*)d_ws + 512 * 1024);       // 256*16384 bf16 (8 MB)
    unsigned short* xbf = (unsigned short*)((char*)d_ws + 512 * 1024 + (KPAD * MDIM * 2));

    const int n_u4 = (KPAD * MDIM * 2) / 16;   // 8MB of zeros
    zero_fill_u4<<<(n_u4 + 255) / 256, 256, 0, stream>>>((uint4*)Wpk, n_u4);

    const int n4 = (16 * 256 * 4096) / 4;      // x elements / 4
    cvt_bf16<<<(n4 + 255) / 256, 256, 0, stream>>>(x, xbf, n4);

    siren_hidden<<<NPTS / 8, 256, 0, stream>>>(gel, W1, b1, W2, b2, h2);
    siren_pack<<<NPTS, 256, 0, stream>>>(gel, W3, b3, h2, Wpk);

    conv_wmma<<<dim3(512, 2, 1), 256, 0, stream>>>(xbf, Wpk, bias, out);
}
